// TwoTowerRecommender_80960133530005
// MI455X (gfx1250) — matrix operand order repaired and checked by reading, not compiled, not run on hardware
//
#include <hip/hip_runtime.h>
#include <hip/hip_bf16.h>
#include <math.h>

// ---------------------------------------------------------------------------
// CDNA5 (gfx1250) two-tower recommender.
// All GEMMs run through v_wmma_f32_16x16x32_f16 (f16 in, f32 accumulate).
// B-fragments are batch-loaded from LDS so WMMAs don't serialize on dscnt.
// ---------------------------------------------------------------------------

typedef __attribute__((ext_vector_type(16))) _Float16 v16h;
typedef __attribute__((ext_vector_type(8)))  _Float16 v8h;
typedef __attribute__((ext_vector_type(8)))  float    v8f;
typedef __attribute__((ext_vector_type(4)))  float    v4f;

union F8  { v8f  v; float    f[8];  };
union H16 { v16h v; _Float16 h[16]; v8h h8[2]; };
union H8  { v8h  v; _Float16 h[8];  };

static constexpr int N_USERS = 100000;
static constexpr int N_TIRES = 200000;
static constexpr int HGT     = 128;
static constexpr int SPEC    = 32;
static constexpr int HIDDEN  = 128;
static constexpr int OUT     = 64;
static constexpr int ITEM_IN = HGT * 3 + SPEC;   // 416
static constexpr int USER_IN = HGT + OUT;        // 192

// f16 element counts of transposed weights in d_ws
static constexpr int SZ_IW1 = HIDDEN * ITEM_IN;  // 53248
static constexpr int SZ_IW2 = HIDDEN * HIDDEN;   // 16384
static constexpr int SZ_IW3 = OUT * HIDDEN;      // 8192
static constexpr int SZ_UW1 = HIDDEN * USER_IN;  // 24576
static constexpr int SZ_UW2 = HIDDEN * HIDDEN;   // 16384
static constexpr int SZ_UW3 = OUT * HIDDEN;      // 8192
static constexpr int SZ_W_ALL = SZ_IW1 + SZ_IW2 + SZ_IW3 + SZ_UW1 + SZ_UW2 + SZ_UW3;

#define LDS_FENCE() asm volatile("s_wait_dscnt 0x0" ::: "memory")

// ---------------------------------------------------------------------------
// Weight prep: WT[n*K + k] = (f16) W[k*N + n]   (W is K x N row-major)
// ---------------------------------------------------------------------------
__global__ void wt_transpose_f16(const float* __restrict__ W,
                                 _Float16* __restrict__ WT, int K, int N) {
    int i = blockIdx.x * 256 + threadIdx.x;
    if (i >= K * N) return;
    int n = i / K, k = i - n * K;
    WT[i] = (_Float16)W[k * N + n];
}

// ---------------------------------------------------------------------------
// WMMA helpers: one 16x16 fp32 tile accumulated over K in chunks of 32.
// A frag: lane half h holds row m=lane&15, K-run of 16 at kc*32 + 16h
// B frag: lane holds col n=lane&15(+tile), K-run of 16 at kc*32 + 16*(lane>>4)
// ---------------------------------------------------------------------------
__device__ __forceinline__ v8f wmma_f16(const v16h& a, const v16h& b, const v8f& c) {
    return __builtin_amdgcn_wmma_f32_16x16x32_f16(false, a, false, b, (short)0, c,
                                                  false, false);
}

__device__ __forceinline__ H8 cvt8(const float* __restrict__ src) {
    v4f a = *(const v4f*)src;
    v4f b = *(const v4f*)(src + 4);
    H8 r;
    r.h[0] = (_Float16)a.x; r.h[1] = (_Float16)a.y;
    r.h[2] = (_Float16)a.z; r.h[3] = (_Float16)a.w;
    r.h[4] = (_Float16)b.x; r.h[5] = (_Float16)b.y;
    r.h[6] = (_Float16)b.z; r.h[7] = (_Float16)b.w;
    return r;
}

// ---------------------------------------------------------------------------
// Item tower: 8 waves/block, 16 tires per wave tile.
// ---------------------------------------------------------------------------
__global__ __launch_bounds__(256) void item_tower_kernel(
    const float* __restrict__ h_tire, const float* __restrict__ h_brand,
    const float* __restrict__ h_size, const float* __restrict__ specs,
    const int* __restrict__ bidx, const int* __restrict__ sidx,
    const _Float16* __restrict__ wsW,  // iW1T | iW2T | iW3T contiguous
    const float* __restrict__ b1, const float* __restrict__ b2,
    const float* __restrict__ b3, float* __restrict__ item_out) {
    // LDS: weights (152 KB) + per-wave activation staging (32 KB)
    __shared__ __align__(32) _Float16 smem[SZ_IW1 + SZ_IW2 + SZ_IW3 + 8 * 16 * HIDDEN];
    _Float16* sW1 = smem;
    _Float16* sW2 = sW1 + SZ_IW1;
    _Float16* sW3 = sW2 + SZ_IW2;

    // cooperative weight load (f16 already, 16B vectors)
    {
        const uint4* src = (const uint4*)wsW;
        uint4* dst = (uint4*)smem;
        const int nv = (SZ_IW1 + SZ_IW2 + SZ_IW3) / 8;  // f16 per uint4 = 8
        for (int i = threadIdx.x; i < nv; i += 256) dst[i] = src[i];
    }
    __syncthreads();

    const int wave = threadIdx.x >> 5;
    const int lane = threadIdx.x & 31;
    const int m    = lane & 15;
    const int half = lane >> 4;
    const int tileBase = (blockIdx.x * 8 + wave) * 16;
    if (tileBase >= N_TIRES) return;

    _Float16* hbuf = smem + SZ_IW1 + SZ_IW2 + SZ_IW3 + wave * 16 * HIDDEN;

    const int t  = tileBase + m;
    const int bi = bidx[t];
    const int si = sidx[t];
    const float* rowT = h_tire + (size_t)t * HGT;
    const float* rowB = h_brand + (size_t)bi * HGT;
    const float* rowS = h_size + (size_t)si * HGT;
    const float* rowP = specs + (size_t)t * SPEC;

    auto loadX8 = [&](int k) -> H8 {
        const float* src;
        if (k < 128)      src = rowT + k;
        else if (k < 256) src = rowB + (k - 128);
        else if (k < 384) src = rowS + (k - 256);
        else              src = rowP + (k - 384);
        return cvt8(src);
    };

    // ---- layer 1: [16 x 416] @ [416 x 128] ----
    F8 acc[8];
#pragma unroll
    for (int nt = 0; nt < 8; nt++)
#pragma unroll
        for (int e = 0; e < 8; e++) acc[nt].f[e] = 0.0f;

    for (int kc = 0; kc < ITEM_IN / 32; kc++) {
        const int k0 = kc * 32 + half * 16;
        H16 af;
        af.h8[0] = loadX8(k0).v;
        af.h8[1] = loadX8(k0 + 8).v;
        H16 bf[8];
#pragma unroll
        for (int nt = 0; nt < 8; nt++)
            bf[nt].v = *(const v16h*)(sW1 + (size_t)(nt * 16 + m) * ITEM_IN +
                                      kc * 32 + half * 16);
#pragma unroll
        for (int nt = 0; nt < 8; nt++)
            acc[nt].v = wmma_f16(af.v, bf[nt].v, acc[nt].v);
    }
    // bias + relu -> LDS (row-major [16][128] f16)
#pragma unroll
    for (int nt = 0; nt < 8; nt++) {
        const int n = nt * 16 + m;
        const float bv = b1[n];
#pragma unroll
        for (int e = 0; e < 8; e++) {
            float v = acc[nt].f[e] + bv;
            v = v > 0.0f ? v : 0.0f;
            hbuf[(e + half * 8) * HIDDEN + n] = (_Float16)v;
        }
    }
    LDS_FENCE();

    // ---- layer 2: [16 x 128] @ [128 x 128] ----
    F8 acc2[8];
#pragma unroll
    for (int nt = 0; nt < 8; nt++)
#pragma unroll
        for (int e = 0; e < 8; e++) acc2[nt].f[e] = 0.0f;

#pragma unroll
    for (int kc = 0; kc < HIDDEN / 32; kc++) {
        const int kb = kc * 32 + half * 16;
        H16 af;
        af.h8[0] = *(const v8h*)(hbuf + m * HIDDEN + kb);
        af.h8[1] = *(const v8h*)(hbuf + m * HIDDEN + kb + 8);
        H16 bf[8];
#pragma unroll
        for (int nt = 0; nt < 8; nt++)
            bf[nt].v = *(const v16h*)(sW2 + (size_t)(nt * 16 + m) * HIDDEN +
                                      kc * 32 + half * 16);
#pragma unroll
        for (int nt = 0; nt < 8; nt++)
            acc2[nt].v = wmma_f16(af.v, bf[nt].v, acc2[nt].v);
    }
    LDS_FENCE();
#pragma unroll
    for (int nt = 0; nt < 8; nt++) {
        const int n = nt * 16 + m;
        const float bv = b2[n];
#pragma unroll
        for (int e = 0; e < 8; e++) {
            float v = acc2[nt].f[e] + bv;
            v = v > 0.0f ? v : 0.0f;
            hbuf[(e + half * 8) * HIDDEN + n] = (_Float16)v;
        }
    }
    LDS_FENCE();

    // ---- layer 3: [16 x 128] @ [128 x 64] ----
    F8 acc3[4];
#pragma unroll
    for (int nt = 0; nt < 4; nt++)
#pragma unroll
        for (int e = 0; e < 8; e++) acc3[nt].f[e] = 0.0f;

#pragma unroll
    for (int kc = 0; kc < HIDDEN / 32; kc++) {
        const int kb = kc * 32 + half * 16;
        H16 af;
        af.h8[0] = *(const v8h*)(hbuf + m * HIDDEN + kb);
        af.h8[1] = *(const v8h*)(hbuf + m * HIDDEN + kb + 8);
        H16 bf[4];
#pragma unroll
        for (int nt = 0; nt < 4; nt++)
            bf[nt].v = *(const v16h*)(sW3 + (size_t)(nt * 16 + m) * HIDDEN +
                                      kc * 32 + half * 16);
#pragma unroll
        for (int nt = 0; nt < 4; nt++)
            acc3[nt].v = wmma_f16(af.v, bf[nt].v, acc3[nt].v);
    }
#pragma unroll
    for (int nt = 0; nt < 4; nt++) {
        const float bv = b3[nt * 16 + m];
#pragma unroll
        for (int e = 0; e < 8; e++) acc3[nt].f[e] += bv;
    }

    // ---- l2 normalize rows (row = e + 8*half), reduce over 16 lanes of half
    float ss[8];
#pragma unroll
    for (int e = 0; e < 8; e++) {
        float s = 0.0f;
#pragma unroll
        for (int nt = 0; nt < 4; nt++) s += acc3[nt].f[e] * acc3[nt].f[e];
        ss[e] = s;
    }
#pragma unroll
    for (int mask = 1; mask <= 8; mask <<= 1)
#pragma unroll
        for (int e = 0; e < 8; e++) ss[e] += __shfl_xor(ss[e], mask, 32);

    float inv[8];
#pragma unroll
    for (int e = 0; e < 8; e++) inv[e] = 1.0f / fmaxf(sqrtf(ss[e]), 1e-12f);

#pragma unroll
    for (int nt = 0; nt < 4; nt++)
#pragma unroll
        for (int e = 0; e < 8; e++) {
            const size_t row = (size_t)(tileBase + half * 8 + e);
            item_out[row * OUT + nt * 16 + m] = acc3[nt].f[e] * inv[e];
        }
}

// ---------------------------------------------------------------------------
// Segment-mean pooling: one wave per user, binary search on sorted segments.
// Deterministic (no float atomics); item table is L2-resident (51 MB).
// ---------------------------------------------------------------------------
__global__ __launch_bounds__(256) void pool_kernel(
    const int* __restrict__ hist_items, const int* __restrict__ hist_segments,
    const float* __restrict__ item_vec, float* __restrict__ pool, int nhist) {
    const int u = blockIdx.x * 8 + (threadIdx.x >> 5);
    if (u >= N_USERS) return;
    const int lane = threadIdx.x & 31;

    auto lower_bound = [&](int key) -> int {
        int lo = 0, hi = nhist;
        while (lo < hi) {
            int mid = (lo + hi) >> 1;
            if (hist_segments[mid] < key) lo = mid + 1; else hi = mid;
        }
        return lo;
    };
    const int s = lower_bound(u);
    const int e = lower_bound(u + 1);

    float a0 = 0.0f, a1 = 0.0f;
    for (int i = s; i < e; i++) {
        if (i + 4 < e) {
            const float* pf = item_vec + (size_t)hist_items[i + 4] * OUT;
            __builtin_prefetch(pf + lane, 0, 2);
        }
        const float* v = item_vec + (size_t)hist_items[i] * OUT;
        a0 += v[lane];
        a1 += v[lane + 32];
    }
    const float c = fmaxf((float)(e - s), 1.0f);
    pool[(size_t)u * OUT + lane]      = a0 / c;
    pool[(size_t)u * OUT + lane + 32] = a1 / c;
}

// ---------------------------------------------------------------------------
// User tower: input = concat(h_user[128], pool[64]) -> 192
// ---------------------------------------------------------------------------
__global__ __launch_bounds__(256) void user_tower_kernel(
    const float* __restrict__ h_user, const float* __restrict__ pool,
    const _Float16* __restrict__ wsW,  // uW1T | uW2T | uW3T contiguous
    const float* __restrict__ b1, const float* __restrict__ b2,
    const float* __restrict__ b3, float* __restrict__ user_out) {
    __shared__ __align__(32) _Float16 smem[SZ_UW1 + SZ_UW2 + SZ_UW3 + 8 * 16 * HIDDEN];
    _Float16* sW1 = smem;
    _Float16* sW2 = sW1 + SZ_UW1;
    _Float16* sW3 = sW2 + SZ_UW2;
    {
        const uint4* src = (const uint4*)wsW;
        uint4* dst = (uint4*)smem;
        const int nv = (SZ_UW1 + SZ_UW2 + SZ_UW3) / 8;
        for (int i = threadIdx.x; i < nv; i += 256) dst[i] = src[i];
    }
    __syncthreads();

    const int wave = threadIdx.x >> 5;
    const int lane = threadIdx.x & 31;
    const int m    = lane & 15;
    const int half = lane >> 4;
    const int tileBase = (blockIdx.x * 8 + wave) * 16;
    if (tileBase >= N_USERS) return;

    _Float16* hbuf = smem + SZ_UW1 + SZ_UW2 + SZ_UW3 + wave * 16 * HIDDEN;

    const int u = tileBase + m;
    const float* rowU = h_user + (size_t)u * HGT;
    const float* rowP = pool + (size_t)u * OUT;

    auto loadX8 = [&](int k) -> H8 {
        const float* src = (k < 128) ? (rowU + k) : (rowP + (k - 128));
        return cvt8(src);
    };

    // ---- layer 1: [16 x 192] @ [192 x 128] ----
    F8 acc[8];
#pragma unroll
    for (int nt = 0; nt < 8; nt++)
#pragma unroll
        for (int e = 0; e < 8; e++) acc[nt].f[e] = 0.0f;

#pragma unroll
    for (int kc = 0; kc < USER_IN / 32; kc++) {
        const int k0 = kc * 32 + half * 16;
        H16 af;
        af.h8[0] = loadX8(k0).v;
        af.h8[1] = loadX8(k0 + 8).v;
        H16 bf[8];
#pragma unroll
        for (int nt = 0; nt < 8; nt++)
            bf[nt].v = *(const v16h*)(sW1 + (size_t)(nt * 16 + m) * USER_IN +
                                      kc * 32 + half * 16);
#pragma unroll
        for (int nt = 0; nt < 8; nt++)
            acc[nt].v = wmma_f16(af.v, bf[nt].v, acc[nt].v);
    }
#pragma unroll
    for (int nt = 0; nt < 8; nt++) {
        const int n = nt * 16 + m;
        const float bv = b1[n];
#pragma unroll
        for (int e = 0; e < 8; e++) {
            float v = acc[nt].f[e] + bv;
            v = v > 0.0f ? v : 0.0f;
            hbuf[(e + half * 8) * HIDDEN + n] = (_Float16)v;
        }
    }
    LDS_FENCE();

    // ---- layer 2 ----
    F8 acc2[8];
#pragma unroll
    for (int nt = 0; nt < 8; nt++)
#pragma unroll
        for (int e = 0; e < 8; e++) acc2[nt].f[e] = 0.0f;
#pragma unroll
    for (int kc = 0; kc < HIDDEN / 32; kc++) {
        const int kb = kc * 32 + half * 16;
        H16 af;
        af.h8[0] = *(const v8h*)(hbuf + m * HIDDEN + kb);
        af.h8[1] = *(const v8h*)(hbuf + m * HIDDEN + kb + 8);
        H16 bf[8];
#pragma unroll
        for (int nt = 0; nt < 8; nt++)
            bf[nt].v = *(const v16h*)(sW2 + (size_t)(nt * 16 + m) * HIDDEN +
                                      kc * 32 + half * 16);
#pragma unroll
        for (int nt = 0; nt < 8; nt++)
            acc2[nt].v = wmma_f16(af.v, bf[nt].v, acc2[nt].v);
    }
    LDS_FENCE();
#pragma unroll
    for (int nt = 0; nt < 8; nt++) {
        const int n = nt * 16 + m;
        const float bv = b2[n];
#pragma unroll
        for (int e = 0; e < 8; e++) {
            float v = acc2[nt].f[e] + bv;
            v = v > 0.0f ? v : 0.0f;
            hbuf[(e + half * 8) * HIDDEN + n] = (_Float16)v;
        }
    }
    LDS_FENCE();

    // ---- layer 3 ----
    F8 acc3[4];
#pragma unroll
    for (int nt = 0; nt < 4; nt++)
#pragma unroll
        for (int e = 0; e < 8; e++) acc3[nt].f[e] = 0.0f;
#pragma unroll
    for (int kc = 0; kc < HIDDEN / 32; kc++) {
        const int kb = kc * 32 + half * 16;
        H16 af;
        af.h8[0] = *(const v8h*)(hbuf + m * HIDDEN + kb);
        af.h8[1] = *(const v8h*)(hbuf + m * HIDDEN + kb + 8);
        H16 bf[4];
#pragma unroll
        for (int nt = 0; nt < 4; nt++)
            bf[nt].v = *(const v16h*)(sW3 + (size_t)(nt * 16 + m) * HIDDEN +
                                      kc * 32 + half * 16);
#pragma unroll
        for (int nt = 0; nt < 4; nt++)
            acc3[nt].v = wmma_f16(af.v, bf[nt].v, acc3[nt].v);
    }
#pragma unroll
    for (int nt = 0; nt < 4; nt++) {
        const float bv = b3[nt * 16 + m];
#pragma unroll
        for (int e = 0; e < 8; e++) acc3[nt].f[e] += bv;
    }

    float ss[8];
#pragma unroll
    for (int e = 0; e < 8; e++) {
        float s = 0.0f;
#pragma unroll
        for (int nt = 0; nt < 4; nt++) s += acc3[nt].f[e] * acc3[nt].f[e];
        ss[e] = s;
    }
#pragma unroll
    for (int mask = 1; mask <= 8; mask <<= 1)
#pragma unroll
        for (int e = 0; e < 8; e++) ss[e] += __shfl_xor(ss[e], mask, 32);

    float inv[8];
#pragma unroll
    for (int e = 0; e < 8; e++) inv[e] = 1.0f / fmaxf(sqrtf(ss[e]), 1e-12f);

#pragma unroll
    for (int nt = 0; nt < 4; nt++)
#pragma unroll
        for (int e = 0; e < 8; e++) {
            const size_t row = (size_t)(tileBase + half * 8 + e);
            user_out[row * OUT + nt * 16 + m] = acc3[nt].f[e] * inv[e];
        }
}

// ---------------------------------------------------------------------------
extern "C" void kernel_launch(void* const* d_in, const int* in_sizes, int n_in,
                              void* d_out, int out_size, void* d_ws, size_t ws_size,
                              hipStream_t stream) {
    const float* h_user    = (const float*)d_in[0];
    const float* h_tire    = (const float*)d_in[1];
    const float* h_brand   = (const float*)d_in[2];
    const float* h_size    = (const float*)d_in[3];
    const float* specs     = (const float*)d_in[4];
    const int*   bidx      = (const int*)d_in[5];
    const int*   sidx      = (const int*)d_in[6];
    const int*   hist_items    = (const int*)d_in[7];
    const int*   hist_segments = (const int*)d_in[8];
    const float* iW1 = (const float*)d_in[9];
    const float* ib1 = (const float*)d_in[10];
    const float* iW2 = (const float*)d_in[11];
    const float* ib2 = (const float*)d_in[12];
    const float* iW3 = (const float*)d_in[13];
    const float* ib3 = (const float*)d_in[14];
    const float* uW1 = (const float*)d_in[15];
    const float* ub1 = (const float*)d_in[16];
    const float* uW2 = (const float*)d_in[17];
    const float* ub2 = (const float*)d_in[18];
    const float* uW3 = (const float*)d_in[19];
    const float* ub3 = (const float*)d_in[20];
    const int nhist = in_sizes[7];

    float* out = (float*)d_out;
    float* user_out = out;
    float* item_out = out + (size_t)N_USERS * OUT;

    _Float16* ws = (_Float16*)d_ws;
    _Float16* iW1T = ws;
    _Float16* iW2T = iW1T + SZ_IW1;
    _Float16* iW3T = iW2T + SZ_IW2;
    _Float16* uW1T = iW3T + SZ_IW3;
    _Float16* uW2T = uW1T + SZ_UW1;
    _Float16* uW3T = uW2T + SZ_UW2;
    float* pool = (float*)(ws + SZ_W_ALL);  // 6.4M floats

    // --- weight transpose + f16 conversion ---
    wt_transpose_f16<<<(SZ_IW1 + 255) / 256, 256, 0, stream>>>(iW1, iW1T, ITEM_IN, HIDDEN);
    wt_transpose_f16<<<(SZ_IW2 + 255) / 256, 256, 0, stream>>>(iW2, iW2T, HIDDEN, HIDDEN);
    wt_transpose_f16<<<(SZ_IW3 + 255) / 256, 256, 0, stream>>>(iW3, iW3T, HIDDEN, OUT);
    wt_transpose_f16<<<(SZ_UW1 + 255) / 256, 256, 0, stream>>>(uW1, uW1T, USER_IN, HIDDEN);
    wt_transpose_f16<<<(SZ_UW2 + 255) / 256, 256, 0, stream>>>(uW2, uW2T, HIDDEN, HIDDEN);
    wt_transpose_f16<<<(SZ_UW3 + 255) / 256, 256, 0, stream>>>(uW3, uW3T, HIDDEN, OUT);

    // --- item tower: 12500 tiles of 16 tires, 8 waves/block ---
    const int item_blocks = (N_TIRES / 16 + 7) / 8;  // 1563
    item_tower_kernel<<<item_blocks, 256, 0, stream>>>(
        h_tire, h_brand, h_size, specs, bidx, sidx, iW1T, ib1, ib2, ib3, item_out);

    // --- history pooling: one wave per user ---
    pool_kernel<<<(N_USERS + 7) / 8, 256, 0, stream>>>(
        hist_items, hist_segments, item_out, pool, nhist);

    // --- user tower: 6250 tiles of 16 users ---
    const int user_blocks = (N_USERS / 16 + 7) / 8;  // 782
    user_tower_kernel<<<user_blocks, 256, 0, stream>>>(
        h_user, pool, uW1T, ub1, ub2, ub3, user_out);

    (void)n_in; (void)out_size; (void)ws_size;
}
